// DecoderLayer_66065186947660
// MI455X (gfx1250) — compile-verified, matrix operand order checked by reading, not run on hardware
//
#include <hip/hip_runtime.h>

// ---------------------------------------------------------------------------
// Transformer decoder layer for MI455X (gfx1250), f16 WMMA + TDM staging.
// B=4, L=1024, D_MODEL=512, H=8, DK=DV=64, D_INNER=256.
// ---------------------------------------------------------------------------

typedef _Float16 f16;
typedef __attribute__((ext_vector_type(16))) _Float16 v16h;
typedef __attribute__((ext_vector_type(8)))  _Float16 v8h;
typedef __attribute__((ext_vector_type(8)))  float    v8f;
typedef __attribute__((ext_vector_type(4)))  unsigned int v4u;
typedef __attribute__((ext_vector_type(8)))  int v8i_t;
typedef __attribute__((ext_vector_type(4)))  int v4i_t;

#define LOG2E 1.44269504088896f

// ---- WMMA helpers ----------------------------------------------------------

__device__ __forceinline__ v8f wmma_f16(v16h a, v16h b, v8f c) {
  // D = A(16x32 f16) * B(32x16 f16) + C(16x16 f32)
  return __builtin_amdgcn_wmma_f32_16x16x32_f16(
      /*neg_a=*/false, a, /*neg_b=*/false, b,
      /*c_mod=*/(short)0, c, /*reuse_a=*/false, /*reuse_b=*/false);
}

// Load a 16x32 fragment from a row-major f16 matrix (A-style layout).
// lane&15 selects the row; lane>>4 selects the K-half (per ISA table).
__device__ __forceinline__ v16h frag_load(const f16* base, int ld, int row0, int k0) {
  const int lane = threadIdx.x & 31;
  const int r  = row0 + (lane & 15);
  const int kb = k0 + ((lane >> 4) << 3);
  const f16* p = base + (size_t)r * ld + kb;
  v8h lo = *(const v8h*)p;          // K = kb .. kb+7
  v8h hi = *(const v8h*)(p + 16);   // K = kb+16 .. kb+23
  v16h out;
#pragma unroll
  for (int i = 0; i < 8; ++i) { out[i] = lo[i]; out[8 + i] = hi[i]; }
  return out;
}

// Same, but the hi 16 K-values may be out of bounds: zero-fill instead of load.
__device__ __forceinline__ v16h frag_load_guard(const f16* base, int ld, int row0,
                                                int k0, bool hiOK) {
  const int lane = threadIdx.x & 31;
  const int r  = row0 + (lane & 15);
  const int kb = k0 + ((lane >> 4) << 3);
  const f16* p = base + (size_t)r * ld + kb;
  v8h lo = *(const v8h*)p;
  v16h out;
#pragma unroll
  for (int i = 0; i < 8; ++i) out[i] = lo[i];
  if (hiOK) {
    v8h hi = *(const v8h*)(p + 16);
#pragma unroll
    for (int i = 0; i < 8; ++i) out[8 + i] = hi[i];
  } else {
#pragma unroll
    for (int i = 0; i < 8; ++i) out[8 + i] = (f16)0.0f;
  }
  return out;
}

// ---- Elementwise: f32 -> f16 cast ------------------------------------------

__global__ __launch_bounds__(256) void cast_f16_kernel(const float* __restrict__ x,
                                                       f16* __restrict__ y, int n) {
  int i = blockIdx.x * 256 + threadIdx.x;
  if (i < n) y[i] = (f16)x[i];
}

// ---- LayerNorm (one wave32 per row of D=512) -> f16 ------------------------

__global__ __launch_bounds__(256) void ln_f16_kernel(const float* __restrict__ x,
                                                     const float* __restrict__ g,
                                                     const float* __restrict__ bt,
                                                     f16* __restrict__ y, int D) {
  const int lane = threadIdx.x & 31, wid = threadIdx.x >> 5;
  const int row = blockIdx.x * 8 + wid;
  const float* xr = x + (size_t)row * D;
  float s = 0.0f;
  for (int i = lane; i < D; i += 32) s += xr[i];
#pragma unroll
  for (int o = 16; o; o >>= 1) s += __shfl_xor(s, o, 32);
  const float mu = s / (float)D;
  float v = 0.0f;
  for (int i = lane; i < D; i += 32) { float d = xr[i] - mu; v += d * d; }
#pragma unroll
  for (int o = 16; o; o >>= 1) v += __shfl_xor(v, o, 32);
  const float rstd = rsqrtf(v / (float)D + 1e-6f);
  f16* yr = y + (size_t)row * D;
  for (int i = lane; i < D; i += 32)
    yr[i] = (f16)((xr[i] - mu) * rstd * g[i] + bt[i]);
}

// ---- Generic WMMA GEMM: C[M,N] = A[M,K] * W[N,K]^T  (+scale/bias/relu/resid)
// Block: 256 threads = 8 waves (4 in M x 2 in N); block tile 256(M) x 64(N).
// The ENTIRE 64xK weight strip is DMA'd into LDS once per block by the
// Tensor Data Mover (TDM), with D# row padding of 4 dwords so the LDS row
// stride is K+8 halfwords (bank-rotated). The k-loop then runs with no
// barriers and no staging waits. A fragments stream from global (L2-resident).
// outMode: 0 = f32 row-major, 1 = f16 row-major, 2 = f16 V-transposed
//          (stores C[m,n] at Vt[((b*8+h)*64+d)*1024 + l], n=(h,d), m=(b,l))

__global__ __launch_bounds__(256) void gemm_wmma_kernel(
    const f16* __restrict__ A, const f16* __restrict__ W,
    int M, int N, int K, float scale,
    const float* __restrict__ bias, int relu,
    const float* __restrict__ resid,
    float* __restrict__ outF, f16* __restrict__ outH, int outMode) {
  __shared__ f16 lw[64 * (512 + 8)];            // 64 rows, stride K+8 (max 520)
  const int t = threadIdx.x;
  const int lane = t & 31, wid = t >> 5;
  const int wm = wid & 3, wn = wid >> 2;        // 4x2 wave grid
  const int mBlk = blockIdx.y * 256, nBlk = blockIdx.x * 64;
  const int ldl = K + 8;                        // LDS row stride (halfwords)

  // --- TDM: DMA W[nBlk .. nBlk+63, 0..K) into LDS with per-row padding -----
  if (wid == 0) {
    const unsigned ldsb = (unsigned)(size_t)(void*)lw;     // LDS byte offset
    const unsigned long long ga =
        (unsigned long long)(size_t)(W + (size_t)nBlk * K);
    const unsigned dim0 = (unsigned)K;          // elements per row
    const unsigned dim1 = 64u;                  // rows in this view
    const unsigned til0 = (unsigned)K;          // tile width  (elements)
    const unsigned til1 = 64u;                  // tile height (rows)
    const unsigned str0 = (unsigned)K;          // row stride  (elements)
    const unsigned padIv = (K == 512) ? 7u : 6u; // pad every 256 / 128 dwords
    const unsigned padAm = 3u;                   // insert 4 dwords (16B)

    v4u g0;
    g0[0] = 1u;                                  // count=1, no gather
    g0[1] = ldsb;                                // lds_addr
    g0[2] = (unsigned)(ga & 0xffffffffu);        // global_addr[31:0]
    g0[3] = (unsigned)((ga >> 32) & 0x01ffffffu) // global_addr[56:32]
            | (2u << 30);                        // type = 2 ("image")
    v8i_t g1;
    g1[0] = (int)((1u << 16)                     // data_size = 2 bytes
                  | (1u << 20)                   // pad_enable
                  | (padIv << 22) | (padAm << 25));
    g1[1] = (int)((dim0 & 0xffffu) << 16);       // tensor_dim0 lo (bits 79:48)
    g1[2] = (int)((dim0 >> 16) | ((dim1 & 0xffffu) << 16));  // dim0 hi | dim1 lo
    g1[3] = (int)((dim1 >> 16) | (til0 << 16));  // dim1 hi | tile_dim0
    g1[4] = (int)(til1 & 0xffffu);               // tile_dim1 (tile_dim2 = 0)
    g1[5] = (int)str0;                           // tensor_dim0_stride[31:0]
    g1[6] = 0;                                   // stride hi | dim1_stride lo
    g1[7] = 0;
    v4i_t g2 = {0, 0, 0, 0};
    v4i_t g3 = {0, 0, 0, 0};
    v8i_t g4 = {0, 0, 0, 0, 0, 0, 0, 0};         // 6-arg toolchain variant
    __builtin_amdgcn_tensor_load_to_lds(g0, g1, g2, g3, g4, 0);
    __builtin_amdgcn_s_wait_tensorcnt(0);
  }
  __syncthreads();

  // --- Main loop: no barriers, compiler-pipelined ---------------------------
  v8f acc[4][2] = {};
  for (int k0 = 0; k0 < K; k0 += 32) {
    v16h wf0 = frag_load(lw, ldl, wn * 32, k0);
    v16h wf1 = frag_load(lw, ldl, wn * 32 + 16, k0);
#pragma unroll
    for (int i = 0; i < 4; ++i) {
      v16h af = frag_load(A, K, mBlk + wm * 64 + i * 16, k0);
      acc[i][0] = wmma_f16(af, wf0, acc[i][0]);
      acc[i][1] = wmma_f16(af, wf1, acc[i][1]);
    }
  }

#pragma unroll
  for (int i = 0; i < 4; ++i) {
#pragma unroll
    for (int j = 0; j < 2; ++j) {
      const int ncol = nBlk + wn * 32 + j * 16 + (lane & 15);
#pragma unroll
      for (int v = 0; v < 8; ++v) {
        const int mrow = mBlk + wm * 64 + i * 16 + v + ((lane >> 4) << 3);
        float val = acc[i][j][v] * scale;
        if (bias)  val += bias[ncol];
        if (relu)  val = fmaxf(val, 0.0f);
        if (resid) val += resid[(size_t)mrow * N + ncol];
        if (outMode == 0) {
          outF[(size_t)mrow * N + ncol] = val;
        } else if (outMode == 1) {
          outH[(size_t)mrow * N + ncol] = (f16)val;
        } else {  // V-transpose store for attention
          const int h = ncol >> 6, d = ncol & 63;
          const int bb = mrow >> 10, ls = mrow & 1023;
          outH[((size_t)(bb * 8 + h) * 64 + d) * 1024 + ls] = (f16)val;
        }
      }
    }
  }
}

// ---- Flash-style attention (one wave per 16-query tile) --------------------
// Computes S^T = K * Q^T so the exp'd probabilities feed the PV WMMA's
// B-operand with no cross-lane movement. V is pre-transposed [b,h,d,l].
// Q already scaled by 1/sqrt(dk). Output f16 [B*L, 512] at col h*64+d.

__global__ __launch_bounds__(256) void attn_wmma_kernel(
    const f16* __restrict__ Q, const f16* __restrict__ Km,
    const f16* __restrict__ Vt, f16* __restrict__ O, int causal) {
  const int lane = threadIdx.x & 31, wid = threadIdx.x >> 5;
  const int bh = blockIdx.y, b = bh >> 3, h = bh & 7;
  const int q0 = (blockIdx.x * 8 + wid) * 16;

  const f16* qbase = Q  + (size_t)b * 1024 * 512 + h * 64;
  const f16* kbase = Km + (size_t)b * 1024 * 512 + h * 64;
  const f16* vbase = Vt + (size_t)(b * 8 + h) * 64 * 1024;

  const v16h bq0 = frag_load(qbase, 512, q0, 0);   // queries, d 0..31
  const v16h bq1 = frag_load(qbase, 512, q0, 32);  // queries, d 32..63

  v8f o[4] = {};
  float run_max = -1e30f, run_sum = 0.0f;
  const int mglob = q0 + (lane & 15);
  const int nEnd = causal ? (q0 + 16) : 1024;

  for (int n0 = 0; n0 < nEnd; n0 += 32) {
    const bool t1 = (n0 + 16) < nEnd;   // wave-uniform
    v8f s0 = {}, s1;
    {
      v16h ka0 = frag_load(kbase, 512, n0, 0);
      v16h ka1 = frag_load(kbase, 512, n0, 32);
      s0 = wmma_f16(ka0, bq0, s0);
      s0 = wmma_f16(ka1, bq1, s0);
    }
    if (t1) {
      v8f z = {};
      v16h kb0 = frag_load(kbase, 512, n0 + 16, 0);
      v16h kb1 = frag_load(kbase, 512, n0 + 16, 32);
      z = wmma_f16(kb0, bq0, z);
      z = wmma_f16(kb1, bq1, z);
      s1 = z;
    } else {
#pragma unroll
      for (int v = 0; v < 8; ++v) s1[v] = -1e30f;
    }

    if (causal) {
      const int nb = n0 + ((lane >> 4) << 3);
#pragma unroll
      for (int v = 0; v < 8; ++v) {
        if (nb + v > mglob) s0[v] = -1e30f;
        if (t1 && (nb + 16 + v > mglob)) s1[v] = -1e30f;
      }
    }

    // Online softmax per query column (lane pair l, l^16 share column m).
    float lmax = -1e30f;
#pragma unroll
    for (int v = 0; v < 8; ++v) lmax = fmaxf(lmax, fmaxf(s0[v], s1[v]));
    lmax = fmaxf(lmax, __shfl_xor(lmax, 16, 32));
    const float nmax = fmaxf(run_max, lmax);
    const float fac = exp2f((run_max - nmax) * LOG2E);

    float lsum = 0.0f;
    v16h pf;
#pragma unroll
    for (int v = 0; v < 8; ++v) {
      float p0 = exp2f((s0[v] - nmax) * LOG2E);
      float p1 = exp2f((s1[v] - nmax) * LOG2E);
      lsum += p0 + p1;
      pf[v] = (f16)p0;      // B-frag K = khalf*8 + v   == tile0 row layout
      pf[8 + v] = (f16)p1;  // B-frag K = 16+khalf*8+v  == tile1 row layout
    }
    lsum += __shfl_xor(lsum, 16, 32);
    run_sum = run_sum * fac + lsum;
    run_max = nmax;

#pragma unroll
    for (int d4 = 0; d4 < 4; ++d4) {
#pragma unroll
      for (int v = 0; v < 8; ++v) o[d4][v] *= fac;
    }
#pragma unroll
    for (int d4 = 0; d4 < 4; ++d4) {
      v16h va = frag_load_guard(vbase, 1024, d4 * 16, n0, t1);
      o[d4] = wmma_f16(va, pf, o[d4]);
    }
  }

  const float inv = 1.0f / run_sum;
  const int mrow = b * 1024 + q0 + (lane & 15);
  f16* obase = O + (size_t)mrow * 512 + h * 64;
#pragma unroll
  for (int d4 = 0; d4 < 4; ++d4) {
#pragma unroll
    for (int v = 0; v < 8; ++v) {
      const int d = d4 * 16 + v + ((lane >> 4) << 3);
      obase[d] = (f16)(o[d4][v] * inv);
    }
  }
}

// ---------------------------------------------------------------------------

extern "C" void kernel_launch(void* const* d_in, const int* in_sizes, int n_in,
                              void* d_out, int out_size, void* d_ws, size_t ws_size,
                              hipStream_t stream) {
  (void)in_sizes; (void)n_in; (void)out_size; (void)ws_size;

  constexpr int Bsz = 4, L = 1024, D = 512, DI = 256;
  constexpr int M4 = Bsz * L;              // 4096 rows
  constexpr size_t NA = (size_t)M4 * D;    // activation elements

  // Inputs in setup_inputs() dict order.
  const float* dec     = (const float*)d_in[0];
  const float* enc     = (const float*)d_in[1];
  //            mask   =               d_in[2]  (causal handled analytically)
  const float* slf_Wq  = (const float*)d_in[3];
  const float* slf_Wk  = (const float*)d_in[4];
  const float* slf_Wv  = (const float*)d_in[5];
  const float* slf_Wfc = (const float*)d_in[6];
  const float* slf_g   = (const float*)d_in[7];
  const float* slf_b   = (const float*)d_in[8];
  const float* enc_Wq  = (const float*)d_in[9];
  const float* enc_Wk  = (const float*)d_in[10];
  const float* enc_Wv  = (const float*)d_in[11];
  const float* enc_Wfc = (const float*)d_in[12];
  const float* enc_g   = (const float*)d_in[13];
  const float* enc_b   = (const float*)d_in[14];
  const float* ffn_W1  = (const float*)d_in[15];
  const float* ffn_b1  = (const float*)d_in[16];
  const float* ffn_W2  = (const float*)d_in[17];
  const float* ffn_b2  = (const float*)d_in[18];
  const float* ffn_g   = (const float*)d_in[19];
  const float* ffn_b   = (const float*)d_in[20];
  float* out = (float*)d_out;

  // Workspace carve (~50 MB).
  char* wp = (char*)d_ws;
  auto carve = [&](size_t bytes) -> void* {
    void* r = (void*)wp; wp += (bytes + 255) & ~(size_t)255; return r;
  };
  f16* decH  = (f16*)carve(NA * 2);
  f16* encH  = (f16*)carve(NA * 2);
  f16* lnA   = (f16*)carve(NA * 2);
  f16* qbuf  = (f16*)carve(NA * 2);
  f16* kbuf  = (f16*)carve(NA * 2);
  f16* vtbuf = (f16*)carve(NA * 2);
  f16* atbuf = (f16*)carve(NA * 2);
  float* x1  = (float*)carve(NA * 4);
  float* x2  = (float*)carve(NA * 4);
  f16* hbuf  = (f16*)carve((size_t)M4 * DI * 2);
  f16* wH[10];
  for (int i = 0; i < 8; ++i) wH[i] = (f16*)carve((size_t)D * D * 2);
  wH[8] = (f16*)carve((size_t)DI * D * 2);   // ffn W1 [256,512]
  wH[9] = (f16*)carve((size_t)D * DI * 2);   // ffn W2 [512,256]

  auto cast = [&](const float* src, f16* dst, int n) {
    cast_f16_kernel<<<dim3(n / 256), dim3(256), 0, stream>>>(src, dst, n);
  };
  auto layernorm = [&](const float* x, const float* g, const float* b, f16* y) {
    ln_f16_kernel<<<dim3(M4 / 8), dim3(256), 0, stream>>>(x, g, b, y, D);
  };
  auto gemm = [&](const f16* A, const f16* W, int M, int N, int K, float scale,
                  const float* bias, int relu, const float* resid,
                  float* oF, f16* oH, int mode) {
    gemm_wmma_kernel<<<dim3(N / 64, M / 256), dim3(256), 0, stream>>>(
        A, W, M, N, K, scale, bias, relu, resid, oF, oH, mode);
  };
  auto attn = [&](const f16* Q, const f16* K, const f16* V, f16* O, int causal) {
    attn_wmma_kernel<<<dim3(L / 128, Bsz * 8), dim3(256), 0, stream>>>(
        Q, K, V, O, causal);
  };

  // --- Stage 0: precision conversion --------------------------------------
  cast(dec, decH, (int)NA);
  cast(enc, encH, (int)NA);
  const float* wsrc[10] = {slf_Wq, slf_Wk, slf_Wv, slf_Wfc,
                           enc_Wq, enc_Wk, enc_Wv, enc_Wfc, ffn_W1, ffn_W2};
  const int wn[10] = {D*D, D*D, D*D, D*D, D*D, D*D, D*D, D*D, DI*D, D*DI};
  for (int i = 0; i < 10; ++i) cast(wsrc[i], wH[i], wn[i]);

  // --- Stage 1: masked self-attention --------------------------------------
  layernorm(dec, slf_g, slf_b, lnA);
  gemm(lnA,  wH[0], M4, D, D, 0.125f, nullptr, 0, nullptr, nullptr, qbuf, 1); // Q/temp
  gemm(decH, wH[1], M4, D, D, 1.0f,   nullptr, 0, nullptr, nullptr, kbuf, 1); // K
  gemm(decH, wH[2], M4, D, D, 1.0f,   nullptr, 0, nullptr, nullptr, vtbuf, 2);// V^T
  attn(qbuf, kbuf, vtbuf, atbuf, /*causal=*/1);
  gemm(atbuf, wH[3], M4, D, D, 1.0f, nullptr, 0, dec, x1, nullptr, 0);        // +resid

  // --- Stage 2: cross attention --------------------------------------------
  layernorm(x1, enc_g, enc_b, lnA);
  gemm(lnA,  wH[4], M4, D, D, 0.125f, nullptr, 0, nullptr, nullptr, qbuf, 1);
  gemm(encH, wH[5], M4, D, D, 1.0f,   nullptr, 0, nullptr, nullptr, kbuf, 1);
  gemm(encH, wH[6], M4, D, D, 1.0f,   nullptr, 0, nullptr, nullptr, vtbuf, 2);
  attn(qbuf, kbuf, vtbuf, atbuf, /*causal=*/0);
  gemm(atbuf, wH[7], M4, D, D, 1.0f, nullptr, 0, x1, x2, nullptr, 0);

  // --- Stage 3: FFN ---------------------------------------------------------
  layernorm(x2, ffn_g, ffn_b, lnA);
  gemm(lnA,  wH[8], M4, DI, D, 1.0f, ffn_b1, 1, nullptr, nullptr, hbuf, 1);   // ReLU
  gemm(hbuf, wH[9], M4, D, DI, 1.0f, ffn_b2, 0, x2, out, nullptr, 0);         // -> d_out
}
